// TARigPrimaryJointPredNet_19945828123270
// MI455X (gfx1250) — matrix operand order repaired and checked by reading, not compiled, not run on hardware
//
#include <hip/hip_runtime.h>
#include <hip/hip_bf16.h>
#include <stdint.h>

// ---------------------------------------------------------------------------
// Types for CDNA5 WMMA (wave32, 16x16x32 f16 -> f32)
// ---------------------------------------------------------------------------
typedef _Float16 half_t;
typedef __attribute__((ext_vector_type(16))) _Float16 v16h;
typedef __attribute__((ext_vector_type(8)))  _Float16 v8h;
typedef __attribute__((ext_vector_type(8)))  float    v8f;

// ---------------------------------------------------------------------------
// WMMA fragment helpers (layouts per CDNA5 ISA 7.12.2, wave32)
// A (16x32 f16): lane L holds row M = L%16; element i holds
//   K = (i/8)*16 + (L/16)*8 + (i%8)  -> two contiguous 8-half runs per lane.
// B (32x16 f16): lane L holds col N = L%16; element i holds K = (i/8)*16 + (L/16)*8 + (i%8), as for A.
//   Weights are pre-swizzled fragment-major: [kt][nt][lane][16] contiguous.
// C/D (16x16 f32): lane L, vgpr r -> M = (L/16)*8 + r, N = L%16
// ---------------------------------------------------------------------------
__device__ __forceinline__ v16h frag_a_lds(const half_t* A, int lda, int lane) {
  const half_t* row = A + (lane & 15) * lda + ((lane >> 4) << 3);
  v8h lo = *(const v8h*)row;        // K = kb..kb+7
  v8h hi = *(const v8h*)(row + 16); // K = kb+16..kb+23
  return __builtin_shufflevector(lo, hi, 0, 1, 2, 3, 4, 5, 6, 7, 8, 9, 10, 11,
                                 12, 13, 14, 15);
}

__device__ __forceinline__ v16h frag_b_sw(const half_t* __restrict__ Wsw,
                                          int NT, int kt, int nt, bool valid,
                                          int lane) {
  if (!valid) {
    v8h z = {};
    return __builtin_shufflevector(z, z, 0, 1, 2, 3, 4, 5, 6, 7, 8, 9, 10, 11,
                                   12, 13, 14, 15);
  }
  const half_t* p = Wsw + ((((long)kt * NT + nt) << 5) + lane) * 16;
  v8h lo = *(const v8h*)p;
  v8h hi = *(const v8h*)(p + 8);
  return __builtin_shufflevector(lo, hi, 0, 1, 2, 3, 4, 5, 6, 7, 8, 9, 10, 11,
                                 12, 13, 14, 15);
}

__device__ __forceinline__ v8f wmma32(v16h a, v16h b, v8f c) {
  return __builtin_amdgcn_wmma_f32_16x16x32_f16(false, a, false, b, (short)0, c,
                                                false, false);
}

// One k-step of WMMA over this wave's TPW tiles. Abase points at the staged
// 64-row chunk (lda = row stride in halves).
template <int TPW, int NT>
__device__ __forceinline__ void mm_step(const half_t* Abase, int lda,
                                        const half_t* __restrict__ Wsw, int kt,
                                        int wave, int lane, v8f* acc) {
#pragma unroll
  for (int i = 0; i < TPW; ++i) {
    int t = wave * TPW + i;
    int mt = t & 3, nt = t >> 2;
    v16h a = frag_a_lds(Abase + mt * 16 * lda, lda, lane);
    acc[i] = wmma32(a, frag_b_sw(Wsw, NT, kt, nt, true, lane), acc[i]);
  }
}

// ---------------------------------------------------------------------------
// Dense node GEMM: C[M, Nreal] = act(A @ Wsw + bias)
// A: single dense matrix [M, lda], f32 (AF32) or f16, zero-padded to Kpad.
// Block 256 thr = 8 waves, 64x64 output tile, 2 accum tiles/wave.
// Software-pipelined staging: chunk k+1 is loaded to registers while WMMA
// consumes chunk k from LDS (loads can't be hoisted across barriers).
// Assumes M % 64 == 0 (true here: N = 32768).
// ---------------------------------------------------------------------------
template <bool AF32>
__global__ __launch_bounds__(256)
void node_gemm_kernel(const void* __restrict__ Av, long lda,
                      const half_t* __restrict__ Wsw, int Kpad,
                      const float* __restrict__ bias, int Nreal, int Npad,
                      half_t* __restrict__ out16, float* __restrict__ out32,
                      int ldOut, int doRelu) {
  __shared__ __attribute__((aligned(16))) half_t As[64 * 32];
  const int tid  = threadIdx.x;
  const int lane = tid & 31;
  const int wave = tid >> 5;
  const int m0   = blockIdx.x * 64;
  const int n0b  = blockIdx.y * 64;
  const int NT   = Npad >> 4;

  const int mt   = wave & 3;
  const int ntA  = wave >> 2;
  const int rowA = tid >> 2;
  const int part = tid & 3;

  const int nt0 = (n0b >> 4) + ntA;
  const int nt1 = (n0b >> 4) + ntA + 2;
  const bool v0 = nt0 < NT, v1 = nt1 < NT;

  v8f zero = {};
  v8f acc0 = zero, acc1 = zero;

  const long rbase = (long)(m0 + rowA) * lda + part * 8;
  half_t* dp = As + rowA * 32 + part * 8;

  auto load_chunk = [&](int k0) -> v8h {
    if (AF32) {
      const float* p = (const float*)Av + rbase + k0;
      float4 f0 = *(const float4*)p;
      float4 f1 = *(const float4*)(p + 4);
      v8h v;
      v[0] = (half_t)f0.x; v[1] = (half_t)f0.y;
      v[2] = (half_t)f0.z; v[3] = (half_t)f0.w;
      v[4] = (half_t)f1.x; v[5] = (half_t)f1.y;
      v[6] = (half_t)f1.z; v[7] = (half_t)f1.w;
      return v;
    } else {
      return *(const v8h*)((const half_t*)Av + rbase + k0);
    }
  };

  v8h reg = load_chunk(0);
  for (int k0 = 0; k0 < Kpad; k0 += 32) {
    __syncthreads();          // previous chunk's readers done
    *(v8h*)dp = reg;
    __syncthreads();
    // prefetch next chunk (clamped, stays in-bounds) before consuming LDS
    int kn = (k0 + 32 < Kpad) ? (k0 + 32) : k0;
    reg = load_chunk(kn);
    v16h a = frag_a_lds(As + mt * 16 * 32, 32, lane);
    int kt = k0 >> 5;
    acc0 = wmma32(a, frag_b_sw(Wsw, NT, kt, nt0, v0, lane), acc0);
    acc1 = wmma32(a, frag_b_sw(Wsw, NT, kt, nt1, v1, lane), acc1);
  }

  const int mrow  = m0 + mt * 16 + ((lane >> 4) << 3);
  const int ncol0 = n0b + ntA * 16 + (lane & 15);
  const int ncol1 = n0b + (ntA + 2) * 16 + (lane & 15);
#pragma unroll
  for (int r = 0; r < 8; ++r) {
    int m = mrow + r;
    if (ncol0 < Nreal) {
      float v = acc0[r] + bias[ncol0];
      if (doRelu) v = fmaxf(v, 0.f);
      if (out16) out16[(long)m * ldOut + ncol0] = (half_t)v;
      if (out32) out32[(long)m * ldOut + ncol0] = v;
    }
    if (ncol1 < Nreal) {
      float v = acc1[r] + bias[ncol1];
      if (doRelu) v = fmaxf(v, 0.f);
      if (out16) out16[(long)m * ldOut + ncol1] = (half_t)v;
      if (out32) out32[(long)m * ldOut + ncol1] = v;
    }
  }
}

// ---------------------------------------------------------------------------
// Fused EdgeConv (per 64-edge tile), templated on hidden width H and input
// width DIN. The gather chunk loader is branch-free vector loads for
// DIN ∈ {64, 256} (uniform scalar branch on k0 only); scalar path for DIN=6.
//   H = relu([x_dst, x_src - x_dst] @ W1 + b1)   (WMMA, LDS-resident)
//   O = relu(H @ W2 + b2)                         (WMMA)
//   agg[dst, colOff + n] max=  via u32 atomicMax (O >= 0, bit-monotonic)
// Assumes E % 64 == 0 for index loads (true here); scatter still guarded.
// ---------------------------------------------------------------------------
template <int H, int DIN>
__global__ __launch_bounds__(256)
void edgeconv_kernel(const half_t* __restrict__ x,
                     const int* __restrict__ esrc, const int* __restrict__ edst,
                     int E, const half_t* __restrict__ W1sw,
                     const float* __restrict__ b1,
                     const half_t* __restrict__ W2sw,
                     const float* __restrict__ b2, float* __restrict__ agg,
                     int aggld, int colOff) {
  constexpr int TPW = H / 32;                  // accumulator tiles per wave
  constexpr int NT  = H / 16;                  // B tiles in N
  constexpr int XC  = (DIN < 8) ? 8 : DIN;     // storage row stride
  constexpr int KP  = (2 * DIN + 31) & ~31;    // padded K
  constexpr int NCH = KP / 32;                 // 32-wide chunks
  __shared__ __attribute__((aligned(16))) half_t As[64 * 32];
  __shared__ __attribute__((aligned(16))) half_t Hs[64 * H];
  const int tid  = threadIdx.x;
  const int lane = tid & 31;
  const int wave = tid >> 5;
  const int e0   = blockIdx.x * 64;

  const int rowA = tid >> 2;
  const int part = tid & 3;
  const int myE  = e0 + rowA;
  const int sIdx = esrc[myE];
  const int dIdx = edst[myE];
  const half_t* xs = x + (long)sIdx * XC;
  const half_t* xd = x + (long)dIdx * XC;
  half_t* dp = As + rowA * 32 + part * 8;

  auto load_chunk = [&](int k0) -> v8h {
    if (DIN >= 8) {
      if (k0 < DIN) {                       // x_dst columns
        return *(const v8h*)(xd + k0 + part * 8);
      } else {                              // x_src - x_dst columns
        int c = k0 - DIN + part * 8;
        return *(const v8h*)(xs + c) - *(const v8h*)(xd + c);
      }
    } else {                                // DIN = 6: scalar boundary path
      v8h val = {};
      const int gk0 = part * 8;
#pragma unroll
      for (int i = 0; i < 8; ++i) {
        int gk = gk0 + i;
        half_t v = (half_t)0;
        if (gk < DIN) v = xd[gk];
        else if (gk < 2 * DIN) v = (half_t)(xs[gk - DIN] - xd[gk - DIN]);
        val[i] = v;
      }
      return val;
    }
  };

  v8f zero = {};
  v8f acc[TPW];
#pragma unroll
  for (int i = 0; i < TPW; ++i) acc[i] = zero;

  // ---- Phase 1: H1 = relu(A @ W1 + b1), software-pipelined staging ----
  v8h reg = load_chunk(0);
  for (int j = 0; j < NCH; ++j) {
    __syncthreads();
    *(v8h*)dp = reg;
    __syncthreads();
    int kn = (j + 1 < NCH) ? (j + 1) * 32 : j * 32;  // clamped prefetch
    reg = load_chunk(kn);
    mm_step<TPW, NT>(As, 32, W1sw, j, wave, lane, acc);
  }

  __syncthreads();
#pragma unroll
  for (int i = 0; i < TPW; ++i) {
    int t = wave * TPW + i;
    int mt = t & 3, nt = t >> 2;
    int n  = nt * 16 + (lane & 15);
    int mb = mt * 16 + ((lane >> 4) << 3);
#pragma unroll
    for (int r = 0; r < 8; ++r) {
      float v = acc[i][r] + b1[n];
      Hs[(mb + r) * H + n] = (half_t)fmaxf(v, 0.f);
    }
    acc[i] = zero;
  }
  __syncthreads();

  // ---- Phase 2: O = relu(H1 @ W2 + b2) ----
  for (int k0 = 0; k0 < H; k0 += 32)
    mm_step<TPW, NT>(Hs + k0, H, W2sw, k0 >> 5, wave, lane, acc);

  // ---- Scatter: segment-max at dst ----
#pragma unroll
  for (int i = 0; i < TPW; ++i) {
    int t = wave * TPW + i;
    int mt = t & 3, nt = t >> 2;
    int n  = nt * 16 + (lane & 15);
    int mb = mt * 16 + ((lane >> 4) << 3);
#pragma unroll
    for (int r = 0; r < 8; ++r) {
      int e = e0 + mb + r;
      if (e < E) {
        float v = fmaxf(acc[i][r] + b2[n], 0.f);
        int node = edst[e];
        atomicMax((unsigned int*)&agg[(long)node * aggld + colOff + n],
                  __float_as_uint(v));
      }
    }
  }
}

// ---------------------------------------------------------------------------
// Weight conversion: f32 [K, Nn] -> f16 fragment-major [Kpad/32][Npad/16][32][16]
// with optional insertion of `ins_cnt` zero rows at padded row `ins_at`.
// ---------------------------------------------------------------------------
__global__ void convert_swz_kernel(const float* __restrict__ W,
                                   half_t* __restrict__ Wp, int K, int Nn,
                                   int Npad, long total, int ins_at,
                                   int ins_cnt) {
  long id = (long)blockIdx.x * blockDim.x + threadIdx.x;
  if (id >= total) return;
  int i    = (int)(id & 15);
  int lane = (int)((id >> 4) & 31);
  long t   = id >> 9;
  int NT   = Npad >> 4;
  int nt   = (int)(t % NT);
  int kt   = (int)(t / NT);
  int k = kt * 32 + ((i >> 3) << 4) + ((lane >> 4) << 3) + (i & 7);
  int n = nt * 16 + (lane & 15);
  int ko;
  if (k < ins_at) ko = k;
  else if (k < ins_at + ins_cnt) ko = -1;
  else ko = k - ins_cnt;
  Wp[id] = (ko >= 0 && ko < K && n < Nn) ? (half_t)W[(long)ko * Nn + n]
                                         : (half_t)0;
}

__global__ void build_x0_kernel(const float* __restrict__ pos,
                                const float* __restrict__ nrm,
                                half_t* __restrict__ x0, int N) {
  int i = blockIdx.x * blockDim.x + threadIdx.x;
  if (i >= N * 8) return;
  int n = i >> 3, c = i & 7;
  float v = 0.f;
  if (c < 3) v = pos[n * 3 + c];
  else if (c < 6) v = nrm[n * 3 + (c - 3)];
  x0[i] = (half_t)v;
}

// Fill cat[N,1888] cols [1024,1888): [x0(8) | x1(64) | x2(256) | x3(512) | 0(24)]
__global__ void build_cat_feats_kernel(const half_t* __restrict__ x0,
                                       const half_t* __restrict__ x1,
                                       const half_t* __restrict__ x2,
                                       const half_t* __restrict__ x3,
                                       half_t* __restrict__ cat, int N) {
  long i = (long)blockIdx.x * blockDim.x + threadIdx.x;  // 8-wide chunk id
  long total = (long)N * 108;
  if (i >= total) return;
  int row = (int)(i / 108);
  int c   = (int)(i % 108) * 8;  // relative to col 1024
  v8h v;
  if (c < 8)        v = *(const v8h*)(x0 + (long)row * 8 + c);
  else if (c < 72)  v = *(const v8h*)(x1 + (long)row * 64 + (c - 8));
  else if (c < 328) v = *(const v8h*)(x2 + (long)row * 256 + (c - 72));
  else if (c < 840) v = *(const v8h*)(x3 + (long)row * 512 + (c - 328));
  else              v = (v8h){};
  *(v8h*)(cat + (long)row * 1888 + 1024 + c) = v;
}

// Fill cat[N,1888] cols [0,1024) with xg[batch[row]] (broadcast per graph).
__global__ void bcast_xg_kernel(const half_t* __restrict__ xg,
                                const int* __restrict__ batch,
                                half_t* __restrict__ cat, int N) {
  long i = (long)blockIdx.x * blockDim.x + threadIdx.x;  // 8-wide chunk id
  long total = (long)N * 128;
  if (i >= total) return;
  int row = (int)(i >> 7);
  int c   = (int)(i & 127) * 8;
  int b   = batch[row];
  *(v8h*)(cat + (long)row * 1888 + c) = *(const v8h*)(xg + (long)b * 1024 + c);
}

__global__ __launch_bounds__(256)
void pool_max_kernel(const half_t* __restrict__ x4, half_t* __restrict__ xg,
                     int npg, int C) {
  int b = blockIdx.x;
  int c = blockIdx.y * 256 + threadIdx.x;
  if (c >= C) return;
  float m = 0.f;  // clamp(>=0) built in; x4 already >= 0
  const half_t* p = x4 + (long)b * npg * C + c;
  for (int n = 0; n < npg; ++n) m = fmaxf(m, (float)p[(long)n * C]);
  xg[b * C + c] = (half_t)m;
}

__device__ __forceinline__ float blk_red_max(float v, float* red) {
  int tid = threadIdx.x;
  red[tid] = v; __syncthreads();
  for (int s = 128; s > 0; s >>= 1) {
    if (tid < s) red[tid] = fmaxf(red[tid], red[tid + s]);
    __syncthreads();
  }
  float r = red[0]; __syncthreads();
  return r;
}
__device__ __forceinline__ float blk_red_sum(float v, float* red) {
  int tid = threadIdx.x;
  red[tid] = v; __syncthreads();
  for (int s = 128; s > 0; s >>= 1) {
    if (tid < s) red[tid] += red[tid + s];
    __syncthreads();
  }
  float r = red[0]; __syncthreads();
  return r;
}

// One block per (graph, joint): softmax over nodes + attn-weighted pos sum.
__global__ __launch_bounds__(256)
void attn_pool_kernel(const float* __restrict__ logits, int ldl,
                      const float* __restrict__ pos, float* __restrict__ joints,
                      float* __restrict__ attn, int npg, int J) {
  __shared__ float red[256];
  int b = blockIdx.x / J, j = blockIdx.x % J;
  int tid = threadIdx.x;
  const float* lg = logits + (long)b * npg * ldl + j;

  float m = -3.4e38f;
  for (int n = tid; n < npg; n += 256) m = fmaxf(m, lg[(long)n * ldl]);
  float M = blk_red_max(m, red);

  float s = 0.f;
  for (int n = tid; n < npg; n += 256) s += __expf(lg[(long)n * ldl] - M);
  float inv = 1.f / blk_red_sum(s, red);

  const float* pp = pos + (long)b * npg * 3;
  float px = 0.f, py = 0.f, pz = 0.f;
  for (int n = tid; n < npg; n += 256) {
    float a = __expf(lg[(long)n * ldl] - M) * inv;
    attn[((long)b * npg + n) * J + j] = a;
    px += a * pp[n * 3 + 0];
    py += a * pp[n * 3 + 1];
    pz += a * pp[n * 3 + 2];
  }
  float rx = blk_red_sum(px, red);
  float ry = blk_red_sum(py, red);
  float rz = blk_red_sum(pz, red);
  if (tid == 0) {
    joints[(b * J + j) * 3 + 0] = rx;
    joints[(b * J + j) * 3 + 1] = ry;
    joints[(b * J + j) * 3 + 2] = rz;
  }
}

// ---------------------------------------------------------------------------
// Host launcher
// ---------------------------------------------------------------------------
extern "C" void kernel_launch(void* const* d_in, const int* in_sizes, int n_in,
                              void* d_out, int out_size, void* d_ws,
                              size_t ws_size, hipStream_t stream) {
  const float* pos = (const float*)d_in[0];
  const float* nrm = (const float*)d_in[1];
  const int* tplE  = (const int*)d_in[2];
  const int* geoE  = (const int*)d_in[3];
  const int* batch = (const int*)d_in[4];

  const int E_tpl = in_sizes[2] / 2;
  const int E_geo = in_sizes[3] / 2;
  const int N = in_sizes[4];
  const int B = 8, J = 24;
  const int NPG = N / B;

  int pi = 5;
  auto nxt = [&]() { return (const float*)d_in[pi++]; };
  const float *g1tW1 = nxt(), *g1tb1 = nxt(), *g1tW2 = nxt(), *g1tb2 = nxt();
  const float *g1gW1 = nxt(), *g1gb1 = nxt(), *g1gW2 = nxt(), *g1gb2 = nxt();
  const float *g1fW = nxt(), *g1fb = nxt();
  const float *g2tW1 = nxt(), *g2tb1 = nxt(), *g2tW2 = nxt(), *g2tb2 = nxt();
  const float *g2gW1 = nxt(), *g2gb1 = nxt(), *g2gW2 = nxt(), *g2gb2 = nxt();
  const float *g2fW = nxt(), *g2fb = nxt();
  const float *g3tW1 = nxt(), *g3tb1 = nxt(), *g3tW2 = nxt(), *g3tb2 = nxt();
  const float *g3gW1 = nxt(), *g3gb1 = nxt(), *g3gW2 = nxt(), *g3gb2 = nxt();
  const float *g3fW = nxt(), *g3fb = nxt();
  const float *glbW = nxt(), *glbb = nxt();
  const float *trW = nxt(), *trb = nxt();

  char* base = (char*)d_ws;
  size_t off = 0;
  auto alloc = [&](size_t bytes) -> void* {
    off = (off + 255) & ~(size_t)255;
    void* p = base + off;
    off += bytes;
    return p;
  };
  half_t* g1tW1p = (half_t*)alloc(32 * 32 * 2);
  half_t* g1tW2p = (half_t*)alloc(32 * 32 * 2);
  half_t* g1gW1p = (half_t*)alloc(32 * 32 * 2);
  half_t* g1gW2p = (half_t*)alloc(32 * 32 * 2);
  half_t* g1fWp  = (half_t*)alloc(64 * 64 * 2);
  half_t* g2tW1p = (half_t*)alloc(128 * 128 * 2);
  half_t* g2tW2p = (half_t*)alloc(128 * 128 * 2);
  half_t* g2gW1p = (half_t*)alloc(128 * 128 * 2);
  half_t* g2gW2p = (half_t*)alloc(128 * 128 * 2);
  half_t* g2fWp  = (half_t*)alloc(256 * 256 * 2);
  half_t* g3tW1p = (half_t*)alloc(512 * 256 * 2);
  half_t* g3tW2p = (half_t*)alloc(256 * 256 * 2);
  half_t* g3gW1p = (half_t*)alloc(512 * 256 * 2);
  half_t* g3gW2p = (half_t*)alloc(256 * 256 * 2);
  half_t* g3fWp  = (half_t*)alloc(512 * 512 * 2);
  half_t* glbWp  = (half_t*)alloc((size_t)832 * 1024 * 2);
  half_t* trWp   = (half_t*)alloc((size_t)1888 * 32 * 2);
  half_t* x0  = (half_t*)alloc((size_t)N * 8 * 2);
  half_t* x1  = (half_t*)alloc((size_t)N * 64 * 2);
  half_t* x2  = (half_t*)alloc((size_t)N * 256 * 2);
  half_t* x3  = (half_t*)alloc((size_t)N * 512 * 2);
  half_t* x4  = (half_t*)alloc((size_t)N * 1024 * 2);
  half_t* xg  = (half_t*)alloc((size_t)B * 1024 * 2);
  half_t* cat = (half_t*)alloc((size_t)N * 1888 * 2);
  float* agg    = (float*)alloc((size_t)N * 512 * 4);
  float* logits = (float*)alloc((size_t)N * 32 * 4);

  auto padW = [&](const float* W, half_t* Wp, int K, int Nn, int Kp, int Np,
                  int ins_at = 1 << 30, int ins_cnt = 0) {
    long total = (long)Kp * Np;
    convert_swz_kernel<<<(unsigned)((total + 255) / 256), 256, 0, stream>>>(
        W, Wp, K, Nn, Np, total, ins_at, ins_cnt);
  };
  padW(g1tW1, g1tW1p, 12, 32, 32, 32);
  padW(g1tW2, g1tW2p, 32, 32, 32, 32);
  padW(g1gW1, g1gW1p, 12, 32, 32, 32);
  padW(g1gW2, g1gW2p, 32, 32, 32, 32);
  padW(g1fW, g1fWp, 64, 64, 64, 64);
  padW(g2tW1, g2tW1p, 128, 128, 128, 128);
  padW(g2tW2, g2tW2p, 128, 128, 128, 128);
  padW(g2gW1, g2gW1p, 128, 128, 128, 128);
  padW(g2gW2, g2gW2p, 128, 128, 128, 128);
  padW(g2fW, g2fWp, 256, 256, 256, 256);
  padW(g3tW1, g3tW1p, 512, 256, 512, 256);
  padW(g3tW2, g3tW2p, 256, 256, 256, 256);
  padW(g3gW1, g3gW1p, 512, 256, 512, 256);
  padW(g3gW2, g3gW2p, 256, 256, 256, 256);
  padW(g3fW, g3fWp, 512, 512, 512, 512);
  padW(glbW, glbWp, 832, 1024, 832, 1024);
  // transform: insert 2 zero rows at padded row 1030 so concat segments
  // [xg 1024][x0 8][x1 64][x2 256][x3 512] are all 8-aligned (1864 -> 1888).
  padW(trW, trWp, 1862, 24, 1888, 32, 1030, 2);

  build_x0_kernel<<<(N * 8 + 255) / 256, 256, 0, stream>>>(pos, nrm, x0, N);

  // ---- GCU1 (din=6 -> h=32, dout=64) ----
  hipMemsetAsync(agg, 0, (size_t)N * 64 * 4, stream);
  edgeconv_kernel<32, 6><<<E_tpl / 64, 256, 0, stream>>>(
      x0, tplE, tplE + E_tpl, E_tpl, g1tW1p, g1tb1, g1tW2p, g1tb2, agg, 64, 0);
  edgeconv_kernel<32, 6><<<E_geo / 64, 256, 0, stream>>>(
      x0, geoE, geoE + E_geo, E_geo, g1gW1p, g1gb1, g1gW2p, g1gb2, agg, 64, 32);
  node_gemm_kernel<true><<<dim3(N / 64, 1), 256, 0, stream>>>(
      agg, 64, g1fWp, 64, g1fb, 64, 64, x1, nullptr, 64, 1);

  // ---- GCU2 (din=64 -> h=128, dout=256) ----
  hipMemsetAsync(agg, 0, (size_t)N * 256 * 4, stream);
  edgeconv_kernel<128, 64><<<E_tpl / 64, 256, 0, stream>>>(
      x1, tplE, tplE + E_tpl, E_tpl, g2tW1p, g2tb1, g2tW2p, g2tb2, agg, 256, 0);
  edgeconv_kernel<128, 64><<<E_geo / 64, 256, 0, stream>>>(
      x1, geoE, geoE + E_geo, E_geo, g2gW1p, g2gb1, g2gW2p, g2gb2, agg, 256,
      128);
  node_gemm_kernel<true><<<dim3(N / 64, 4), 256, 0, stream>>>(
      agg, 256, g2fWp, 256, g2fb, 256, 256, x2, nullptr, 256, 1);

  // ---- GCU3 (din=256 -> h=256, dout=512) ----
  hipMemsetAsync(agg, 0, (size_t)N * 512 * 4, stream);
  edgeconv_kernel<256, 256><<<E_tpl / 64, 256, 0, stream>>>(
      x2, tplE, tplE + E_tpl, E_tpl, g3tW1p, g3tb1, g3tW2p, g3tb2, agg, 512, 0);
  edgeconv_kernel<256, 256><<<E_geo / 64, 256, 0, stream>>>(
      x2, geoE, geoE + E_geo, E_geo, g3gW1p, g3gb1, g3gW2p, g3gb2, agg, 512,
      256);
  node_gemm_kernel<true><<<dim3(N / 64, 8), 256, 0, stream>>>(
      agg, 512, g3fWp, 512, g3fb, 512, 512, x3, nullptr, 512, 1);

  // ---- concat features into cat[N,1888] cols 1024.. ----
  {
    long total = (long)N * 108;
    build_cat_feats_kernel<<<(unsigned)((total + 255) / 256), 256, 0, stream>>>(
        x0, x1, x2, x3, cat, N);
  }

  // ---- glb: x4 = relu([x1,x2,x3] @ Wg + bg): A = cat cols 1032..1863 ----
  node_gemm_kernel<false><<<dim3(N / 64, 16), 256, 0, stream>>>(
      cat + 1032, 1888, glbWp, 832, glbb, 1024, 1024, x4, nullptr, 1024, 1);

  // ---- per-graph max pool (clamped >= 0) ----
  pool_max_kernel<<<dim3(B, 4), 256, 0, stream>>>(x4, xg, NPG, 1024);

  // ---- broadcast xg into cat cols 0..1023 ----
  {
    long total = (long)N * 128;
    bcast_xg_kernel<<<(unsigned)((total + 255) / 256), 256, 0, stream>>>(
        xg, batch, cat, N);
  }

  // ---- transform: logits = cat @ Wt + bt (no relu) ----
  node_gemm_kernel<false><<<dim3(N / 64, 1), 256, 0, stream>>>(
      cat, 1888, trWp, 1888, trb, 24, 32, nullptr, logits, 32, 0);

  // ---- softmax attention pooling -> d_out = [joints | attn] ----
  float* joints = (float*)d_out;
  float* attn   = joints + (size_t)B * J * 3;
  attn_pool_kernel<<<B * J, 256, 0, stream>>>(logits, 32, pos, joints, attn,
                                              NPG, J);
}
